// VectorQuantizer_13984413516272
// MI455X (gfx1250) — matrix operand order repaired and checked by reading, not compiled, not run on hardware
//
#include <hip/hip_runtime.h>
#include <hip/hip_bf16.h>

typedef __attribute__((ext_vector_type(16))) __bf16 v16bf;
typedef __attribute__((ext_vector_type(8)))  __bf16 v8bf;
typedef __attribute__((ext_vector_type(8)))  float  v8f;

// ---------------- problem sizes ----------------
#define CH      256          // channels == CODEBOOK_DIM
#define NROWS   8192         // 8 * 32 * 32 tokens
#define KCODES  8192         // codebook entries
#define DDIM    256

// ---------------- d_out layout (floats, concatenated in return order) ----
#define ZQ_OFF   0                       // [8,256,32,32] = 2097152
#define LOSS_OFF 2097152
#define PERP_OFF 2097153
#define ENC_OFF  2097154                 // [8192,8192] one-hot
#define NENC     67108864ull
#define IDX_OFF  69206018                // [8192] indices as float

// ---------------- main GEMM tiling ----------------
#define THREADS   128                    // 4 waves (wave32)
#define ROWS_WG   64                     // 16 z-rows per wave
#define CPP       64                     // codes staged per phase
#define NPHASE    (KCODES / CPP)
#define DSTRIDE   272                    // 256 + 16 bf16 pad -> 544B rows (32B aligned)
#define SMEM_MAIN ((size_t)(2*ROWS_WG*DSTRIDE + 2*CPP*DSTRIDE)*2 + CPP*sizeof(float))

// ---------------- workspace layout (bytes) ----------------
#define WS_IDX    0
#define WS_CNT    32768
#define WS_ENORM  65536
#define WS_EHI    98304
#define WS_ELO    (98304 + 4194304)
#define WS_NEED   (98304 + 2*4194304)

// fp32 -> bf16 hi/lo split (truncation; residual ~2^-16 relative)
__device__ __forceinline__ void bf16split(float v, unsigned short& h, unsigned short& l) {
    unsigned u  = __float_as_uint(v);
    unsigned hu = u & 0xFFFF0000u;
    h = (unsigned short)(hu >> 16);
    float rem = v - __uint_as_float(hu);
    l = (unsigned short)(__float_as_uint(rem) >> 16);
}

// CDNA5 async copy: 16B global -> LDS, tracked by ASYNCcnt (no VGPR staging)
__device__ __forceinline__ void async_cp16(unsigned lds_off, const void* g) {
    asm volatile("global_load_async_to_lds_b128 %0, %1, off"
                 :: "v"(lds_off), "v"(g) : "memory");
}
__device__ __forceinline__ void wait_async0() {
    asm volatile("s_wait_asynccnt 0x0" ::: "memory");
}

union V16 { v16bf v; v8bf h[2]; };

__device__ __forceinline__ void loadA(V16& ah, V16& al,
        const unsigned short* ah_p, const unsigned short* al_p, int kk) {
    ah.h[0] = *(const v8bf*)(ah_p + kk * 32);
    ah.h[1] = *(const v8bf*)(ah_p + kk * 32 + 16);
    al.h[0] = *(const v8bf*)(al_p + kk * 32);
    al.h[1] = *(const v8bf*)(al_p + kk * 32 + 16);
}

// ---------------- init: zero histogram + loss slot ----------------
__global__ void k_init(float* __restrict__ out, int* __restrict__ counts) {
    int t = blockIdx.x * blockDim.x + threadIdx.x;
    for (int k = t; k < KCODES; k += blockDim.x * gridDim.x) counts[k] = 0;
    if (t == 0) out[LOSS_OFF] = 0.0f;
}

// ---------------- zero-fill the 256MB one-hot region ----------------
__global__ void k_fill_enc(float* __restrict__ out) {
    float2* p = (float2*)(out + ENC_OFF);            // ENC_OFF is even -> 8B aligned
    const size_t n2 = NENC / 2;
    const size_t stride = (size_t)blockDim.x * gridDim.x;
    for (size_t i = (size_t)blockIdx.x * blockDim.x + threadIdx.x; i < n2; i += stride)
        p[i] = make_float2(0.0f, 0.0f);
}

// ---------------- codebook row norms (exact fp32) ----------------
__global__ void k_enorm(const float* __restrict__ emb, float* __restrict__ enorm) {
    const int lane = threadIdx.x & 31;
    const int wave = threadIdx.x >> 5;
    const int k = blockIdx.x * 8 + wave;
    const float* e = emb + (size_t)k * DDIM;
    float s = 0.0f;
#pragma unroll
    for (int j = 0; j < 8; ++j) { float v = e[lane + j * 32]; s += v * v; }
    for (int off = 16; off >= 1; off >>= 1) s += __shfl_xor(s, off, 32);
    if (lane == 0) enorm[k] = s;
}

// ---------------- one-time codebook fp32 -> bf16 hi/lo ----------------
__global__ void k_convE(const float* __restrict__ emb,
                        unsigned short* __restrict__ eh,
                        unsigned short* __restrict__ el) {
    int i = blockIdx.x * blockDim.x + threadIdx.x;   // over KCODES*DDIM
    unsigned short h, l; bf16split(emb[i], h, l);
    eh[i] = h; el[i] = l;
}

// ---------------- main: bf16-split WMMA distance GEMM + running argmin ----
__global__ __launch_bounds__(THREADS)
void k_main(const float* __restrict__ z, const float* __restrict__ emb,
            const unsigned short* __restrict__ emb_hi,
            const unsigned short* __restrict__ emb_lo,
            const float* __restrict__ enorm, int* __restrict__ idx_ws,
            int* __restrict__ counts, int preconv) {
    extern __shared__ char smem[];
    unsigned short* zt_hi = (unsigned short*)smem;                 // [ROWS_WG][DSTRIDE]
    unsigned short* zt_lo = zt_hi + ROWS_WG * DSTRIDE;
    unsigned short* ct_hi = zt_lo + ROWS_WG * DSTRIDE;             // [CPP][DSTRIDE]
    unsigned short* ct_lo = ct_hi + CPP * DSTRIDE;
    float* en_s = (float*)(ct_lo + CPP * DSTRIDE);                 // [CPP]

    const int t   = threadIdx.x;
    const int n0  = blockIdx.x * ROWS_WG;       // rows of this WG (one batch image)
    const int b   = n0 >> 10;
    const int hw0 = n0 & 1023;
    const float* zb = z + ((size_t)b << 18);    // b * 256 * 1024

    // ---- stage z tile: NCHW gather -> [row][d] bf16 hi/lo in LDS (coalesced in hw)
#pragma unroll 4
    for (int it = 0; it < (ROWS_WG * DDIM) / THREADS; ++it) {
        int i = t + THREADS * it;
        int r = i & (ROWS_WG - 1);
        int d = i >> 6;
        float v = zb[((size_t)d << 10) + hw0 + r];
        unsigned short h, l; bf16split(v, h, l);
        zt_hi[r * DSTRIDE + d] = h;
        zt_lo[r * DSTRIDE + d] = l;
    }
    __syncthreads();

    const int lane = t & 31;
    const int wave = t >> 5;
    const int half = lane >> 4;
    const int m    = lane & 15;

    // ---- B operand (this wave's 16 z-rows, all of D=256) pinned in VGPRs.
    // B 32x16 bf16 layout: lane holds column n = lane%16; elements 0..7 = K (half?8..15:0..7),
    // elements 8..15 = K (half?24..31:16..23), matching the A operand order.
    V16 bh[8], bl[8];
    {
        const int brow = wave * 16 + m;
        const unsigned short* ph = zt_hi + brow * DSTRIDE + half * 8;
        const unsigned short* pl = zt_lo + brow * DSTRIDE + half * 8;
#pragma unroll
        for (int kk = 0; kk < 8; ++kk) {
            bh[kk].h[0] = *(const v8bf*)(ph + kk * 32);
            bh[kk].h[1] = *(const v8bf*)(ph + kk * 32 + 16);
            bl[kk].h[0] = *(const v8bf*)(pl + kk * 32);
            bl[kk].h[1] = *(const v8bf*)(pl + kk * 32 + 16);
        }
    }

    float best = 3.0e38f;
    int   bidx = 0;

    for (int phase = 0; phase < NPHASE; ++phase) {
        const int cbase = phase * CPP;
        __syncthreads();   // previous phase's compute done before restaging codes

        if (preconv) {
            // ---- pure bf16 copy via CDNA5 async-to-LDS (ASYNCcnt path)
            const unsigned short* ghi = emb_hi + (size_t)cbase * DDIM;
            const unsigned short* glo = emb_lo + (size_t)cbase * DDIM;
#pragma unroll 4
            for (int c = t; c < (CPP * DDIM) / 8; c += THREADS) {   // 16B chunks
                int row = c >> 5, seg = c & 31;
                unsigned l0 = (unsigned)(uintptr_t)(ct_hi + row * DSTRIDE + seg * 8);
                unsigned l1 = (unsigned)(uintptr_t)(ct_lo + row * DSTRIDE + seg * 8);
                async_cp16(l0, ghi + row * DDIM + seg * 8);
                async_cp16(l1, glo + row * DDIM + seg * 8);
            }
            wait_async0();
        } else {
            // ---- fallback: stage fp32 -> bf16 hi/lo in-kernel
            const float* eb = emb + (size_t)cbase * DDIM;
#pragma unroll 4
            for (int it = 0; it < (CPP * DDIM) / THREADS; ++it) {
                int i = t + THREADS * it;
                int c = i >> 8;
                int d = i & 255;
                unsigned short h, l; bf16split(eb[i], h, l);
                ct_hi[c * DSTRIDE + d] = h;
                ct_lo[c * DSTRIDE + d] = l;
            }
        }
        if (t < CPP) en_s[t] = enorm[cbase + t];

        // prefetch next phase's codebook slab into cache (global_prefetch_b8)
        if (phase + 1 < NPHASE) {
            if (preconv) {
                __builtin_prefetch(emb_hi + (size_t)(cbase + CPP) * DDIM + t * 128, 0, 1);
                __builtin_prefetch(emb_lo + (size_t)(cbase + CPP) * DDIM + t * 128, 0, 1);
            } else {
                const float* nb = emb + (size_t)(cbase + CPP) * DDIM;
                __builtin_prefetch(nb + t * 128, 0, 1);
                __builtin_prefetch(nb + t * 128 + 64, 0, 1);
            }
        }
        __syncthreads();

        // ---- 4 sub-tiles of 16 codes; scores^T = codes x z-rows.
        // A 16x32 bf16 layout: lane m holds code row m; half-wave selects K chunk:
        // V0..3 = K (half?8..15:0..7), V4..7 = K (half?24..31:16..23).
#pragma unroll
        for (int sub = 0; sub < CPP / 16; ++sub) {
            const unsigned short* ah_p = ct_hi + (sub * 16 + m) * DSTRIDE + half * 8;
            const unsigned short* al_p = ct_lo + (sub * 16 + m) * DSTRIDE + half * 8;
            v8f acc = {0, 0, 0, 0, 0, 0, 0, 0};
            // 2-deep software pipeline of A tiles so ds_load latency overlaps WMMA
            V16 a0h, a0l, a1h, a1l;
            loadA(a0h, a0l, ah_p, al_p, 0);
#pragma unroll
            for (int kk = 0; kk < 8; ++kk) {
                if ((kk & 1) == 0) {
                    if (kk < 7) loadA(a1h, a1l, ah_p, al_p, kk + 1);
                    // dot(z,e) ~= zh*eh + zl*eh + zh*el  (bf16x3 split-precision)
                    acc = __builtin_amdgcn_wmma_f32_16x16x32_bf16(false, a0h.v, false, bh[kk].v, (short)0, acc, false, false);
                    acc = __builtin_amdgcn_wmma_f32_16x16x32_bf16(false, a0h.v, false, bl[kk].v, (short)0, acc, false, false);
                    acc = __builtin_amdgcn_wmma_f32_16x16x32_bf16(false, a0l.v, false, bh[kk].v, (short)0, acc, false, false);
                } else {
                    if (kk < 7) loadA(a0h, a0l, ah_p, al_p, kk + 1);
                    acc = __builtin_amdgcn_wmma_f32_16x16x32_bf16(false, a1h.v, false, bh[kk].v, (short)0, acc, false, false);
                    acc = __builtin_amdgcn_wmma_f32_16x16x32_bf16(false, a1h.v, false, bl[kk].v, (short)0, acc, false, false);
                    acc = __builtin_amdgcn_wmma_f32_16x16x32_bf16(false, a1l.v, false, bh[kk].v, (short)0, acc, false, false);
                }
            }
            // D layout: lane column = z-row (lane%16); VGPR r = code (r + 8*half)
            const int csub = cbase + sub * 16 + half * 8;
#pragma unroll
            for (int r = 0; r < 8; ++r) {
                float s = en_s[sub * 16 + half * 8 + r] - 2.0f * acc[r];
                if (s < best) { best = s; bidx = csub + r; }
            }
        }
    }

    // merge the two half-wave code partitions for each z-row
    float ob = __shfl_xor(best, 16, 32);
    int   oi = __shfl_xor(bidx, 16, 32);
    if (ob < best || (ob == best && oi < bidx)) { best = ob; bidx = oi; }
    if (half == 0) {
        int n = n0 + wave * 16 + m;
        idx_ws[n] = bidx;
        atomicAdd(&counts[bidx], 1);
    }
}

// ---------------- output pass: z_q (NCHW) + exact fp32 loss ----------------
#define OROWS 32
__global__ void k_out(const float* __restrict__ z, const float* __restrict__ emb,
                      const int* __restrict__ idx_ws, float* __restrict__ out) {
    __shared__ float wsum[8];
    const int t   = threadIdx.x;
    const int n0  = blockIdx.x * OROWS;
    const int b   = n0 >> 10;
    const int hw0 = n0 & 1023;
    const float* zb = z + ((size_t)b << 18);
    float* ob2 = out + ZQ_OFF + ((size_t)b << 18);
    float lsum = 0.0f;
#pragma unroll 4
    for (int it = 0; it < (OROWS * CH) / 256; ++it) {
        int i = t + 256 * it;
        int r = i & (OROWS - 1);
        int d = i >> 5;
        int id = idx_ws[n0 + r];
        float e  = emb[(size_t)id * DDIM + d];
        size_t a = ((size_t)d << 10) + hw0 + r;
        float zv = zb[a];
        ob2[a] = e;                      // straight-through output == emb[idx]
        float df = e - zv;
        lsum += df * df;
    }
    for (int off = 16; off >= 1; off >>= 1) lsum += __shfl_xor(lsum, off, 32);
    if ((t & 31) == 0) wsum[t >> 5] = lsum;
    __syncthreads();
    if (t == 0) {
        float s = 0.0f;
        for (int w = 0; w < 8; ++w) s += wsum[w];
        // loss = (BETA + 1) * mean((z_q - z)^2)
        atomicAdd(&out[LOSS_OFF], s * (1.25f / 2097152.0f));
    }
}

// ---------------- one-hot scatter + indices ----------------
__global__ void k_scatter(const int* __restrict__ idx_ws, float* __restrict__ out) {
    int n = blockIdx.x * 256 + threadIdx.x;
    if (n < NROWS) {
        int id = idx_ws[n];
        out[ENC_OFF + (size_t)n * KCODES + id] = 1.0f;
        out[IDX_OFF + n] = (float)id;
    }
}

// ---------------- perplexity ----------------
__global__ void k_perp(const int* __restrict__ counts, float* __restrict__ out) {
    __shared__ float wsum[8];
    float s = 0.0f;
    for (int k = threadIdx.x; k < KCODES; k += 256) {
        float p = (float)counts[k] * (1.0f / 8192.0f);
        s += p * logf(p + 1e-10f);
    }
    for (int off = 16; off >= 1; off >>= 1) s += __shfl_xor(s, off, 32);
    if ((threadIdx.x & 31) == 0) wsum[threadIdx.x >> 5] = s;
    __syncthreads();
    if (threadIdx.x == 0) {
        float tot = 0.0f;
        for (int w = 0; w < 8; ++w) tot += wsum[w];
        out[PERP_OFF] = expf(-tot);
    }
}

extern "C" void kernel_launch(void* const* d_in, const int* in_sizes, int n_in,
                              void* d_out, int out_size, void* d_ws, size_t ws_size,
                              hipStream_t stream) {
    (void)in_sizes; (void)n_in; (void)out_size;
    const float* z   = (const float*)d_in[0];   // [8,256,32,32] fp32
    const float* emb = (const float*)d_in[1];   // [8192,256] fp32
    float* out = (float*)d_out;

    int*   idx_ws = (int*)((char*)d_ws + WS_IDX);      // [8192]
    int*   counts = (int*)((char*)d_ws + WS_CNT);      // [8192]
    float* enorm  = (float*)((char*)d_ws + WS_ENORM);  // [8192]
    unsigned short* emb_hi = (unsigned short*)((char*)d_ws + WS_EHI);
    unsigned short* emb_lo = (unsigned short*)((char*)d_ws + WS_ELO);
    const int preconv = (ws_size >= (size_t)WS_NEED) ? 1 : 0;   // ws_size is fixed -> deterministic

    hipFuncSetAttribute(reinterpret_cast<const void*>(k_main),
                        hipFuncAttributeMaxDynamicSharedMemorySize, (int)SMEM_MAIN);

    k_init    <<<32, 256, 0, stream>>>(out, counts);
    k_fill_enc<<<2048, 256, 0, stream>>>(out);
    k_enorm   <<<KCODES / 8, 256, 0, stream>>>(emb, enorm);
    if (preconv)
        k_convE <<<(KCODES * DDIM) / 256, 256, 0, stream>>>(emb, emb_hi, emb_lo);
    k_main    <<<NROWS / ROWS_WG, THREADS, SMEM_MAIN, stream>>>(z, emb, emb_hi, emb_lo,
                                                               enorm, idx_ws, counts, preconv);
    k_out     <<<NROWS / OROWS, 256, 0, stream>>>(z, emb, idx_ws, out);
    k_scatter <<<NROWS / 256, 256, 0, stream>>>(idx_ws, out);
    k_perp    <<<1, 256, 0, stream>>>(counts, out);
}
